// S6_88665304858808
// MI455X (gfx1250) — compile-verified, matrix operand order checked by reading, not run on hardware
//
#include <hip/hip_runtime.h>
#include <hip/hip_bf16.h>
#include <math.h>

// ---- model dims (fixed by reference) ----
#define BB    4
#define LSEQ  1024
#define DIN   256          // D_IN
#define DII   512          // EXPAND * D_IN
#define NST   16           // D_STATE
#define DTR   16           // DT_RANK
#define DCONV 4
#define NTOK  (BB*LSEQ)    // 4096 tokens

typedef __attribute__((ext_vector_type(16))) __bf16 v16bf;
typedef __attribute__((ext_vector_type(8)))  float  v8f;

// =====================================================================
// Weight conversion: (batch, K, N) f32 row-major  ->  (batch, N, Kpad) bf16
// (transposed so WMMA B-fragment loads are contiguous; zero-padded in K)
// =====================================================================
__global__ void s6_cvt_wt(const float* __restrict__ in, __bf16* __restrict__ out,
                          int K, int N, int Kpad, int batch)
{
  long total = (long)batch * N * Kpad;
  for (long i = (long)blockIdx.x * blockDim.x + threadIdx.x; i < total;
       i += (long)gridDim.x * blockDim.x) {
    int  k = (int)(i % Kpad);
    long t = i / Kpad;
    int  n = (int)(t % N);
    int  g = (int)(t / N);
    float v = (k < K) ? in[(long)g * K * N + (long)k * N + n] : 0.0f;
    out[i] = (__bf16)v;
  }
}

// =====================================================================
// WMMA GEMM:  C[M,N] = A[M,K] (f32, compile-time lda) @ W[K,N] (+ residual)
// W given as bf16 transposed (N x KPAD).  One wave computes a 16 x (16*NT)
// tile: one A fragment is reused by NT WMMAs per 32-wide K slice.
// All K in {16,32,256,512} are multiples of 8 -> guards fold at compile time.
// =====================================================================
template <int KC, int KPAD, int LDA, int NT, bool HAS_RES>
__global__ __launch_bounds__(32) void s6_wmma_gemm(
    const float*  __restrict__ A,
    const __bf16* __restrict__ Wt,
    const float*  __restrict__ Res,
    float*        __restrict__ C,
    int N)
{
  const int lane  = threadIdx.x;
  const int half  = lane >> 4;            // 0: lanes 0-15, 1: lanes 16-31
  const int l16   = lane & 15;
  const int row   = blockIdx.y * 16 + l16;     // A row (M index)
  const int ncol0 = blockIdx.x * (16 * NT);    // first N column of this wave

  const float* arow = A + (size_t)row * LDA;

  v8f acc[NT] = {};

  for (int kc = 0; kc < KPAD; kc += 32) {
    const int kb0 = kc + half * 8;        // first contiguous 8-run of K
    const int kb1 = kc + 16 + half * 8;   // second contiguous 8-run of K

    // ---- A fragment: fp32 -> bf16 (two 8-element runs per lane) ----
    union { v16bf v; __bf16 e[16]; } af;
    if ((KC & 31) == 0 || kb0 + 8 <= KC) {
      const float4 f0 = *(const float4*)(arow + kb0);
      const float4 f1 = *(const float4*)(arow + kb0 + 4);
      af.e[0] = (__bf16)f0.x; af.e[1] = (__bf16)f0.y;
      af.e[2] = (__bf16)f0.z; af.e[3] = (__bf16)f0.w;
      af.e[4] = (__bf16)f1.x; af.e[5] = (__bf16)f1.y;
      af.e[6] = (__bf16)f1.z; af.e[7] = (__bf16)f1.w;
    } else {
      #pragma unroll
      for (int i = 0; i < 8; ++i) af.e[i] = (__bf16)0.0f;
    }
    if ((KC & 31) == 0 || kb1 + 8 <= KC) {
      const float4 f0 = *(const float4*)(arow + kb1);
      const float4 f1 = *(const float4*)(arow + kb1 + 4);
      af.e[8]  = (__bf16)f0.x; af.e[9]  = (__bf16)f0.y;
      af.e[10] = (__bf16)f0.z; af.e[11] = (__bf16)f0.w;
      af.e[12] = (__bf16)f1.x; af.e[13] = (__bf16)f1.y;
      af.e[14] = (__bf16)f1.z; af.e[15] = (__bf16)f1.w;
    } else {
      #pragma unroll
      for (int i = 0; i < 8; ++i) af.e[8 + i] = (__bf16)0.0f;
    }

    // prefetch next K-slice of A (emits global_prefetch_b8)
    __builtin_prefetch(arow + kc + 32, 0, 0);

    // ---- NT B fragments + WMMAs sharing this A fragment ----
    #pragma unroll
    for (int j = 0; j < NT; ++j) {
      const __bf16* wrow = Wt + (size_t)(ncol0 + j * 16 + l16) * KPAD;
      union { v16bf v; unsigned int u[8]; } bf_;
      *(uint4*)&bf_.u[0] = *(const uint4*)(wrow + kb0);
      *(uint4*)&bf_.u[4] = *(const uint4*)(wrow + kb1);
      acc[j] = __builtin_amdgcn_wmma_f32_16x16x32_bf16(
          false, af.v, false, bf_.v, (short)0, acc[j], false, false);
    }
  }

  // ---- store: VGPR r -> M = r + 8*half, N = l16 (per tile j) ----
  #pragma unroll
  for (int j = 0; j < NT; ++j) {
    const int ncol = ncol0 + j * 16 + l16;
    #pragma unroll
    for (int r = 0; r < 8; ++r) {
      const int m = blockIdx.y * 16 + r + 8 * half;
      float v = acc[j][r];
      if (HAS_RES) v += Res[(size_t)m * N + ncol];
      C[(size_t)m * N + ncol] = v;
    }
  }
}

// =====================================================================
// RMSNorm over rows of DIN=256; one wave per row
// =====================================================================
__global__ __launch_bounds__(256) void s6_rmsnorm(
    const float* __restrict__ X, const float* __restrict__ w,
    float* __restrict__ Y, int rows)
{
  int wave = threadIdx.x >> 5, lane = threadIdx.x & 31;
  int row = blockIdx.x * 8 + wave;
  if (row >= rows) return;
  const float* xr = X + (size_t)row * DIN;
  float v[8], s = 0.f;
  for (int j = 0; j < 8; ++j) { v[j] = xr[lane + 32 * j]; s += v[j] * v[j]; }
  for (int off = 16; off; off >>= 1) s += __shfl_xor(s, off, 32);
  float r = rsqrtf(s * (1.0f / DIN) + 1e-6f);
  float* yr = Y + (size_t)row * DIN;
  for (int j = 0; j < 8; ++j) yr[lane + 32 * j] = v[j] * r * w[lane + 32 * j];
}

// =====================================================================
// Depthwise causal conv (D_CONV=4) + SiLU.  xp from first half of xz.
// =====================================================================
__global__ void s6_conv_silu(const float* __restrict__ xz,
                             const float* __restrict__ convw,
                             const float* __restrict__ convb,
                             float* __restrict__ xp)
{
  int idx = blockIdx.x * blockDim.x + threadIdx.x;  // over BB*LSEQ*DII
  if (idx >= BB * LSEQ * DII) return;
  int d = idx % DII; int t = idx / DII; int l = t % LSEQ; int b = t / LSEQ;
  float acc = convb[d];
  for (int k = 0; k < DCONV; ++k) {
    int ll = l + k - (DCONV - 1);
    if (ll >= 0)
      acc += xz[((size_t)(b * LSEQ + ll)) * (2 * DII) + d] * convw[d * DCONV + k];
  }
  xp[idx] = acc / (1.f + __expf(-acc));   // silu
}

// =====================================================================
// Selective scan, fused: softplus(dt) -> dA=exp(dt*A) on the fly ->
// state update -> y = h.C + Dp*xp -> * silu(z).
// One lane per (b,d) channel, 16 states in registers.
// Per-step B/C broadcast (32 floats) staged via async global->LDS DMA.
// grid = 8 blocks x 256 threads (block = half the channels of one batch)
// =====================================================================
__global__ __launch_bounds__(256) void s6_scan(
    const float* __restrict__ dtp,  const float* __restrict__ bdt,
    const float* __restrict__ xp,   const float* __restrict__ xdbl,
    const float* __restrict__ xz,   const float* __restrict__ Alog,
    const float* __restrict__ Dp,   float* __restrict__ yg)
{
  __shared__ float sBC[32];              // [0..15] = B, [16..31] = C
  const int b = blockIdx.x >> 1;
  const int d = ((blockIdx.x & 1) << 8) + threadIdx.x;

  float A[NST], h[NST];
  for (int n = 0; n < NST; ++n) { A[n] = -__expf(Alog[d * NST + n]); h[n] = 0.f; }
  const float Dv = Dp[d], bd = bdt[d];

  for (int l = 0; l < LSEQ; ++l) {
    const size_t tok = (size_t)b * LSEQ + l;

    // async DMA the 32-float B/C slice of this token straight into LDS
    if (threadIdx.x < 32) {
      unsigned lds_off = (unsigned)(size_t)(&sBC[threadIdx.x]);
      const float* gptr = xdbl + tok * (DTR + 2 * NST) + DTR + threadIdx.x;
      asm volatile("global_load_async_to_lds_b32 %0, %1, off"
                   :: "v"(lds_off), "v"(gptr) : "memory");
    }
    asm volatile("s_wait_asynccnt 0x0" ::: "memory");
    __syncthreads();

    float x  = dtp[tok * DII + d] + bd;
    float dt = (x > 20.f) ? x : __logf(1.f + __expf(x));  // softplus
    float xv = xp[tok * DII + d];
    float du = dt * xv;
    float acc = 0.f;
    #pragma unroll
    for (int n = 0; n < NST; ++n) {
      h[n] = __expf(dt * A[n]) * h[n] + du * sBC[n];
      acc += h[n] * sBC[16 + n];
    }
    float z = xz[tok * (2 * DII) + DII + d];
    float g = z / (1.f + __expf(-z));                      // silu(z)
    yg[tok * DII + d] = (acc + Dv * xv) * g;
    __syncthreads();
  }
}

// =====================================================================
// ctxdot[b] = mean_l ( h0[b,l,:] . Whead[128:384] + h1[b,l,:] . Whead[384:640] )
// =====================================================================
__global__ __launch_bounds__(256) void s6_ctxdot(
    const float* __restrict__ h0, const float* __restrict__ h1,
    const float* __restrict__ Whead, float* __restrict__ ctxdot)
{
  __shared__ float red[256];
  const int b = blockIdx.x;
  float acc = 0.f;
  for (int i = threadIdx.x; i < LSEQ * DIN; i += 256) {
    int c = i & (DIN - 1);
    acc += h0[(size_t)b * LSEQ * DIN + i] * Whead[128 + c]
         + h1[(size_t)b * LSEQ * DIN + i] * Whead[384 + c];
  }
  red[threadIdx.x] = acc;
  __syncthreads();
  for (int s = 128; s; s >>= 1) {
    if (threadIdx.x < s) red[threadIdx.x] += red[threadIdx.x + s];
    __syncthreads();
  }
  if (threadIdx.x == 0) ctxdot[b] = red[0] * (1.0f / LSEQ);
}

// w16[f] = Wemb_trg[f,:] . Whead[:128]   (folds yemb GEMM into the head)
__global__ void s6_w16(const float* __restrict__ Wemb_trg,
                       const float* __restrict__ Whead, float* __restrict__ w16)
{
  int f = threadIdx.x;
  if (f < 16) {
    float a = 0.f;
    for (int e = 0; e < 128; ++e) a += Wemb_trg[f * 128 + e] * Whead[e];
    w16[f] = a;
  }
}

// out[b,t] = y[b,t,:] . w16 + ctxdot[b] + bhead
__global__ void s6_head(const float* __restrict__ y, const float* __restrict__ w16,
                        const float* __restrict__ ctxdot, const float* __restrict__ bhead,
                        float* __restrict__ out)
{
  int i = blockIdx.x * blockDim.x + threadIdx.x;
  if (i >= BB * 256) return;
  int b = i >> 8;
  const float* yr = y + (size_t)i * 16;
  float a = bhead[0] + ctxdot[b];
  for (int f = 0; f < 16; ++f) a += yr[f] * w16[f];
  out[i] = a;
}

// reverse along L
__global__ void s6_reverse(const float* __restrict__ in, float* __restrict__ out)
{
  int i = blockIdx.x * blockDim.x + threadIdx.x;
  if (i >= BB * LSEQ * DIN) return;
  int c = i % DIN; int t = i / DIN; int l = t % LSEQ; int b = t / LSEQ;
  out[((size_t)(b * LSEQ + (LSEQ - 1 - l))) * DIN + c] = in[i];
}

// =====================================================================
extern "C" void kernel_launch(void* const* d_in, const int* in_sizes, int n_in,
                              void* d_out, int out_size, void* d_ws, size_t ws_size,
                              hipStream_t stream)
{
  const float* X        = (const float*)d_in[0];
  const float* yin      = (const float*)d_in[1];
  const float* Wemb_in  = (const float*)d_in[2];
  const float* Wemb_trg = (const float*)d_in[3];
  const float* Win      = (const float*)d_in[4];
  const float* convw    = (const float*)d_in[5];
  const float* convb    = (const float*)d_in[6];
  const float* Wx       = (const float*)d_in[7];
  const float* Wdt      = (const float*)d_in[8];
  const float* bdt      = (const float*)d_in[9];
  const float* Alog     = (const float*)d_in[10];
  const float* Dp       = (const float*)d_in[11];
  const float* Wout     = (const float*)d_in[12];
  const float* norm_w   = (const float*)d_in[13];
  const float* Whead    = (const float*)d_in[14];
  const float* bhead    = (const float*)d_in[15];
  float* out = (float*)d_out;

  char* ws = (char*)d_ws;
  size_t off = 0;
  auto alloc = [&](size_t bytes) -> void* {
    void* p = ws + off;
    off = (off + bytes + 255) & ~(size_t)255;
    return p;
  };

  float* Xemb   = (float*)alloc((size_t)NTOK * DIN * 4);   // dir0 running h
  float* Xrev   = (float*)alloc((size_t)NTOK * DIN * 4);   // dir1 running h
  float* tmpH   = (float*)alloc((size_t)NTOK * DIN * 4);
  float* xn     = (float*)alloc((size_t)NTOK * DIN * 4);
  float* xz     = (float*)alloc((size_t)NTOK * 2 * DII * 4);
  float* xp     = (float*)alloc((size_t)NTOK * DII * 4);
  float* xdbl   = (float*)alloc((size_t)NTOK * (DTR + 2 * NST) * 4);
  float* dtp    = (float*)alloc((size_t)NTOK * DII * 4);
  float* yg     = (float*)alloc((size_t)NTOK * DII * 4);
  __bf16* WembT = (__bf16*)alloc((size_t)DIN * 32 * 2);
  __bf16* WinT  = (__bf16*)alloc((size_t)4 * 2 * DII * DIN * 2);
  __bf16* WxT   = (__bf16*)alloc((size_t)4 * (DTR + 2 * NST) * DII * 2);
  __bf16* WdtT  = (__bf16*)alloc((size_t)4 * DII * 32 * 2);
  __bf16* WoutT = (__bf16*)alloc((size_t)4 * DIN * DII * 2);
  float* w16    = (float*)alloc(16 * 4);
  float* ctxd   = (float*)alloc(4 * 4);
  (void)ws_size; (void)in_sizes; (void)n_in; (void)out_size;

  // ---- weight conversions (transpose + pad + bf16) ----
  s6_cvt_wt<<<64,   256, 0, stream>>>(Wemb_in, WembT, 32, DIN, 32, 1);
  s6_cvt_wt<<<2048, 256, 0, stream>>>(Win,  WinT,  DIN, 2 * DII, DIN, 4);
  s6_cvt_wt<<<512,  256, 0, stream>>>(Wx,   WxT,   DII, DTR + 2 * NST, DII, 4);
  s6_cvt_wt<<<256,  256, 0, stream>>>(Wdt,  WdtT,  DTR, DII, 32, 4);
  s6_cvt_wt<<<1024, 256, 0, stream>>>(Wout, WoutT, DII, DIN, DII, 4);

  // ---- X embedding (WMMA): (4096x32)@(32x256) ----
  s6_wmma_gemm<32, 32, 32, 4, false>
      <<<dim3(DIN / 64, NTOK / 16), 32, 0, stream>>>(X, WembT, nullptr, Xemb, DIN);

  // ---- reversed copy for direction 1 ----
  s6_reverse<<<(BB * LSEQ * DIN + 255) / 256, 256, 0, stream>>>(Xemb, Xrev);

  // ---- bidirectional mamba stack ----
  for (int dir = 0; dir < 2; ++dir) {
    float* hbuf = dir ? Xrev : Xemb;
    for (int l = 0; l < 2; ++l) {
      const int dl = dir * 2 + l;
      const float* hin  = (l == 0) ? hbuf : tmpH;
      float*       hout = (l == 0) ? tmpH : hbuf;

      s6_rmsnorm<<<NTOK / 8, 256, 0, stream>>>(hin, norm_w + dl * DIN, xn, NTOK);

      // xz = xn @ Win : (4096x256)@(256x1024)
      s6_wmma_gemm<256, 256, 256, 4, false>
          <<<dim3((2 * DII) / 64, NTOK / 16), 32, 0, stream>>>(
              xn, WinT + (size_t)dl * 2 * DII * DIN, nullptr, xz, 2 * DII);

      s6_conv_silu<<<(BB * LSEQ * DII + 255) / 256, 256, 0, stream>>>(
          xz, convw + dl * DII * DCONV, convb + dl * DII, xp);

      // xdbl = xp @ Wx : (4096x512)@(512x48), N tiles = 3
      s6_wmma_gemm<512, 512, 512, 3, false>
          <<<dim3(1, NTOK / 16), 32, 0, stream>>>(
              xp, WxT + (size_t)dl * (DTR + 2 * NST) * DII, nullptr, xdbl,
              DTR + 2 * NST);

      // dtp = dt_raw @ Wdt : (4096x16, lda=48)@(16x512), Kpad=32
      s6_wmma_gemm<16, 32, DTR + 2 * NST, 4, false>
          <<<dim3(DII / 64, NTOK / 16), 32, 0, stream>>>(
              xdbl, WdtT + (size_t)dl * DII * 32, nullptr, dtp, DII);

      s6_scan<<<2 * BB, 256, 0, stream>>>(
          dtp, bdt + dl * DII, xp, xdbl, xz,
          Alog + (size_t)dl * DII * NST, Dp + dl * DII, yg);

      // hout = hin + yg @ Wout : (4096x512)@(512x256), fused residual
      s6_wmma_gemm<512, 512, 512, 4, true>
          <<<dim3(DIN / 64, NTOK / 16), 32, 0, stream>>>(
              yg, WoutT + (size_t)dl * DIN * DII, hin, hout, DIN);
    }
  }

  // ---- folded head ----
  s6_ctxdot<<<BB, 256, 0, stream>>>(Xemb, Xrev, Whead, ctxd);
  s6_w16<<<1, 32, 0, stream>>>(Wemb_trg, Whead, w16);
  s6_head<<<(BB * 256 + 255) / 256, 256, 0, stream>>>(yin, w16, ctxd, bhead, out);
}
